// _SCNN_31499290149270
// MI455X (gfx1250) — matrix operand order repaired and checked by reading, not compiled, not run on hardware
//
#include <hip/hip_runtime.h>

typedef __bf16 bf16;
typedef __attribute__((ext_vector_type(16))) __bf16 v16bf;
typedef __attribute__((ext_vector_type(8)))  __bf16 v8bf;
typedef __attribute__((ext_vector_type(8)))  float  v8f;

#define B_   16
#define C_   128
#define H_   72
#define W_   200
#define K_   9
#define HW_  (H_*W_)
#define CHW_ (C_*HW_)

#define CHUNK 32
#define XROWS (CHUNK + 8)   // chunk + 2*PAD halo
#define PITCH 136           // 128 ci + 8 pad (bank stagger: 272B row = 68 dwords)

// --- helpers ---------------------------------------------------------------

__device__ __forceinline__ bf16 f2bf(float f) {
  union { float f; unsigned u; } a; a.f = f;
  unsigned r = a.u + 0x7FFFu + ((a.u >> 16) & 1u);   // round-to-nearest-even
  unsigned short h = (unsigned short)(r >> 16);
  union { unsigned short s; bf16 b; } o; o.s = h; return o.b;
}

__device__ __forceinline__ v16bf ld_v16(const bf16* p) {
  v8bf lo = *(const v8bf*)p;          // ds_load_b128
  v8bf hi = *(const v8bf*)(p + 8);    // ds_load_b128
  v16bf r;
#pragma unroll
  for (int i = 0; i < 8; ++i) { r[i] = lo[i]; r[i + 8] = hi[i]; }
  return r;
}

// --- pack weights into A-fragment (16x32 bf16) per-lane order, contiguous k as in B --------
// packed[((((sweep*8+cot)*9+k)*4+ct)*32+lane)*16+j]
//   lane<16: m=lane,    K = j
//   lane>=16:m=lane-16, K = j+16
__global__ void pack_weights(const float* __restrict__ wd,
                             const float* __restrict__ wu,
                             const float* __restrict__ wr,
                             const float* __restrict__ wl,
                             bf16* __restrict__ packed) {
  int idx = blockIdx.x * 256 + threadIdx.x;
  int j    = idx & 15;
  int lane = (idx >> 4) & 31;
  int ct   = (idx >> 9) & 3;
  int rest = idx >> 11;
  int k    = rest % 9;
  int cot  = (rest / 9) & 7;
  int sweep = idx / (8 * 9 << 11);
  if (sweep >= 4) return;
  const float* w = (sweep == 0) ? wd : (sweep == 1) ? wu : (sweep == 2) ? wr : wl;
  int m  = lane & 15;
  int kl = j + (lane & 16);  // +16 for upper half-wave
  int ci = ct * 32 + kl;
  int co = cot * 16 + m;
  // (C,C,1,K) and (C,C,K,1) both flatten to co*C*K + ci*K + k
  packed[idx] = f2bf(w[(co * C_ + ci) * K_ + k]);
}

// --- one sweep step: out[cur line] += relu( W (x) out[prev line] ) ---------
__global__ __launch_bounds__(256) void sweep_step(
    float* __restrict__ out, const bf16* __restrict__ pw,
    int cur, int prev, int L, int line_stride, int scan_stride) {
  __shared__ bf16 lds[XROWS * PITCH];

  const int tid    = threadIdx.x;
  const int b      = blockIdx.y;
  const int chunk0 = blockIdx.x * CHUNK;

  // Stage prev line (+halo) transposed [x][ci] as bf16, zero-padded at ends.
  const float* src = out + (size_t)b * CHW_ + (size_t)prev * scan_stride;
  for (int i = tid; i < XROWS * C_; i += 256) {
    int x  = i % XROWS;
    int ci = i / XROWS;
    int pos = chunk0 - 4 + x;
    float v = 0.0f;
    if (pos >= 0 && pos < L) v = src[(size_t)ci * HW_ + (size_t)pos * line_stride];
    lds[x * PITCH + ci] = f2bf(v);
  }
  __syncthreads();

  const int lane = tid & 31;
  const int wv   = tid >> 5;       // wave -> co tile (8 waves x 16 co = 128)
  const int n    = lane & 15;      // B/D column within tile
  const int hi16 = (lane & 16);    // upper half-wave -> K += 16

  v8f acc0 = {}; v8f acc1 = {};
  const bf16* pa = pw + (size_t)wv * (9 * 4 * 512);

#pragma unroll
  for (int k = 0; k < K_; ++k) {
#pragma unroll
    for (int ct = 0; ct < 4; ++ct) {
      v16bf a = *(const v16bf*)(pa + (k * 4 + ct) * 512 + lane * 16);
      int cib = ct * 32 + hi16;
      v16bf b0 = ld_v16(&lds[(n + k)      * PITCH + cib]);
      v16bf b1 = ld_v16(&lds[(16 + n + k) * PITCH + cib]);
      acc0 = __builtin_amdgcn_wmma_f32_16x16x32_bf16(false, a, false, b0,
                                                     (short)0, acc0, false, false);
      acc1 = __builtin_amdgcn_wmma_f32_16x16x32_bf16(false, a, false, b1,
                                                     (short)0, acc1, false, false);
    }
  }

  // Epilogue: D layout — VGPR r: M = r + 8*(lane>=16), N = lane%16
  float* oline = out + (size_t)b * CHW_ + (size_t)cur * scan_stride;
  const int mb = (lane >> 4) * 8;
#pragma unroll
  for (int t = 0; t < 2; ++t) {
    int col = chunk0 + t * 16 + n;
    if (col < L) {
      v8f acc = t ? acc1 : acc0;
#pragma unroll
      for (int r = 0; r < 8; ++r) {
        int co = wv * 16 + mb + r;
        float v = acc[r];
        float* p = oline + (size_t)co * HW_ + (size_t)col * line_stride;
        *p += (v > 0.0f ? v : 0.0f);
      }
    }
  }
}

// --- host driver ------------------------------------------------------------

extern "C" void kernel_launch(void* const* d_in, const int* in_sizes, int n_in,
                              void* d_out, int out_size, void* d_ws, size_t ws_size,
                              hipStream_t stream) {
  const float* x  = (const float*)d_in[0];
  const float* wd = (const float*)d_in[1];
  const float* wu = (const float*)d_in[2];
  const float* wr = (const float*)d_in[3];
  const float* wl = (const float*)d_in[4];
  float* out  = (float*)d_out;
  bf16* packed = (bf16*)d_ws;   // 4 * 147456 bf16 = 1.18 MB

  hipMemcpyAsync(out, x, sizeof(float) * (size_t)B_ * CHW_,
                 hipMemcpyDeviceToDevice, stream);

  pack_weights<<<2304, 256, 0, stream>>>(wd, wu, wr, wl, packed);

  const size_t SW = (size_t)8 * 9 * 4 * 512;  // per-sweep packed elements
  dim3 blk(256);
  dim3 grow(7, B_);   // ceil(200/32) x batches
  dim3 gcol(3, B_);   // ceil(72/32)  x batches

  // down sweep over H (line along W)
  for (int h = 1; h < H_; ++h)
    sweep_step<<<grow, blk, 0, stream>>>(out, packed + 0 * SW, h, h - 1, W_, 1, W_);
  // up sweep over H (endpoints preserved)
  for (int h = H_ - 2; h >= 1; --h)
    sweep_step<<<grow, blk, 0, stream>>>(out, packed + 1 * SW, h, h + 1, W_, 1, W_);
  // right sweep over W (line along H)
  for (int w = 1; w < W_; ++w)
    sweep_step<<<gcol, blk, 0, stream>>>(out, packed + 2 * SW, w, w - 1, H_, W_, 1);
  // left sweep over W (endpoints preserved)
  for (int w = W_ - 2; w >= 1; --w)
    sweep_step<<<gcol, blk, 0, stream>>>(out, packed + 3 * SW, w, w + 1, H_, W_, 1);
}
